// GroupedQueryAttention_18751827215051
// MI455X (gfx1250) — matrix operand order repaired and checked by reading, not compiled, not run on hardware
//
#include <hip/hip_runtime.h>

// ---------------------------------------------------------------------------
// GQA + RoPE attention for MI455X (gfx1250), bf16 WMMA pipeline with
// double-buffered GLOBAL_LOAD_ASYNC_TO_LDS staging (ASYNCcnt-tracked).
// ---------------------------------------------------------------------------

typedef __attribute__((ext_vector_type(16))) __bf16 v16bf;
typedef __attribute__((ext_vector_type(8)))  __bf16 v8bf;
typedef __attribute__((ext_vector_type(8)))  float  v8f;
typedef __attribute__((ext_vector_type(4)))  unsigned int u32x4;
typedef __attribute__((ext_vector_type(4)))  int          i32x4;

#define BATCH   2
#define SEQ     2048
#define DMODEL  2048
#define HQ      16
#define HKV     4
#define DHEAD   128
#define GQA_G   (HQ / HKV)
#define MROWS   (BATCH * SEQ)          // 4096

#if defined(__gfx1250__) && __has_builtin(__builtin_amdgcn_global_load_async_to_lds_b128) && \
    __has_builtin(__builtin_amdgcn_s_wait_asynccnt)
#define HAVE_ASYNC_LDS 1
#else
#define HAVE_ASYNC_LDS 0
#endif

// copy 16 bytes global -> LDS (async if available, else sync round trip)
static __device__ inline void cp16_g2l(void* lds_dst, const void* gsrc) {
#if HAVE_ASYNC_LDS
    __builtin_amdgcn_global_load_async_to_lds_b128(
        (__attribute__((address_space(1))) i32x4*)gsrc,
        (__attribute__((address_space(3))) i32x4*)lds_dst, 0, 0);
#else
    *(u32x4*)lds_dst = *(const u32x4*)gsrc;
#endif
}

#if HAVE_ASYNC_LDS
#define ASYNC_WAIT(n) __builtin_amdgcn_s_wait_asynccnt(n)
#else
#define ASYNC_WAIT(n) ((void)0)
#endif

static __device__ inline v16bf cat16(v8bf lo, v8bf hi) {
    return __builtin_shufflevector(lo, hi, 0,1,2,3,4,5,6,7,8,9,10,11,12,13,14,15);
}

static __device__ inline v8f wmma_bf16(v16bf a, v16bf b, v8f c) {
    // D(16x16,f32) = A(16x32,bf16) x B(32x16,bf16) + C
    return __builtin_amdgcn_wmma_f32_16x16x32_bf16(false, a, false, b,
                                                   (short)0, c, false, false);
}

// ---------------------------------------------------------------------------
// Kernel 1: fp32 -> bf16 cast
// ---------------------------------------------------------------------------
__global__ void cast_to_bf16(const float* __restrict__ in,
                             __bf16* __restrict__ out, int n) {
    for (int i = blockIdx.x * blockDim.x + threadIdx.x; i < n;
         i += gridDim.x * blockDim.x)
        out[i] = (__bf16)in[i];
}

// ---------------------------------------------------------------------------
// Kernel 2: WMMA GEMM  C[m,n] = sum_k A[m,k] * W[n,k]   (A:[M,D], W:[N,D])
//   block = 128 rows x 128 cols, 8 waves, wave tile = 16 rows x 128 cols
//   W tile double-buffered in LDS via async global->LDS copies.
//   MODE 0: RoPE epilogue, store bf16 head-major [B, nHeads, S, DHEAD]  (Q,K)
//   MODE 1: store bf16 TRANSPOSED  [B, nHeads, DHEAD, S]                (V)
//   MODE 2: store fp32 row-major   [M, DMODEL]                          (out)
// ---------------------------------------------------------------------------
template <int MODE>
__global__ __launch_bounds__(256, 1)
void gemm_wmma(const __bf16* __restrict__ A, const __bf16* __restrict__ W,
               __bf16* __restrict__ outB, float* __restrict__ outF, int nHeads) {
    __shared__ __bf16 Wl[2][128][32];       // double-buffered W tile (16 KB)

    const int lane = threadIdx.x & 31;
    const int wv   = threadIdx.x >> 5;      // 0..7
    const int l16  = lane & 15;
    const int hl   = lane >> 4;             // lane half (0/1)
    const int m0   = blockIdx.x * 128 + wv * 16;
    const int n0   = blockIdx.y * 128;
    const int rowA = m0 + l16;

    // staging role of this thread: 2 x 16B per tile
    const int sn   = threadIdx.x >> 1;              // W row (col of C)
    const int hseg = (threadIdx.x & 1) * 16;        // k sub-segment

    v8f acc[8];
#pragma unroll
    for (int f = 0; f < 8; ++f) acc[f] = (v8f){0,0,0,0,0,0,0,0};

    // prologue: stage tile 0 into buffer 0
    {
        const __bf16* src = &W[(size_t)(n0 + sn) * DMODEL + hseg];
        cp16_g2l(&Wl[0][sn][hseg],     src);
        cp16_g2l(&Wl[0][sn][hseg + 8], src + 8);
    }

    const int NK = DMODEL / 32;             // 64 k-steps
    for (int i = 0; i < NK; ++i) {
        const int cur = i & 1;
        const int k   = i * 32;
        if (i + 1 < NK) {                   // issue next tile, drain current
            const __bf16* src = &W[(size_t)(n0 + sn) * DMODEL + k + 32 + hseg];
            cp16_g2l(&Wl[cur ^ 1][sn][hseg],     src);
            cp16_g2l(&Wl[cur ^ 1][sn][hseg + 8], src + 8);
            ASYNC_WAIT(2);
        } else {
            ASYNC_WAIT(0);
        }
        __syncthreads();                    // tile i visible to all waves

        // ---- A fragment (16x32 bf16): row = lane%16, K half by lane half ----
        v8bf alo = *(const v8bf*)&A[(size_t)rowA * DMODEL + k + hl * 8];
        v8bf ahi = *(const v8bf*)&A[(size_t)rowA * DMODEL + k + 16 + hl * 8];
        v16bf af = cat16(alo, ahi);
        if (i + 1 < NK)
            __builtin_prefetch(&A[(size_t)rowA * DMODEL + k + 32], 0, 1);

#pragma unroll
        for (int nf = 0; nf < 8; ++nf) {
            // B fragment (32x16): col = lane%16, K 8h..8h+7 / 16+8h..16+8h+7 by lane half h
            v8bf blo = *(const v8bf*)&Wl[cur][nf * 16 + l16][hl * 8];
            v8bf bhi = *(const v8bf*)&Wl[cur][nf * 16 + l16][16 + hl * 8];
            acc[nf]  = wmma_bf16(af, cat16(blo, bhi), acc[nf]);
        }
        __syncthreads();                    // protect buffer before next overwrite
    }

    // ---- epilogues (C layout: value acc[f][r] -> m = r + hl*8, n = f*16+l16) ----
    const int h = n0 / DHEAD;   // one head per 128-col block

    if (MODE == 0) {            // RoPE + head-major bf16 store
#pragma unroll
        for (int f = 0; f < 4; ++f) {
            const int j = f * 16 + l16;                      // 0..63
            const float inv_freq = __expf(-(float)j * (9.210340371976184f / 64.0f));
#pragma unroll
            for (int r = 0; r < 8; ++r) {
                const int m   = m0 + r + hl * 8;
                const int b   = m >> 11;                     // / SEQ
                const int pos = m & (SEQ - 1);
                const float th = (float)pos * inv_freq;
                const float cs = __cosf(th), sn2 = __sinf(th);
                const float q1 = acc[f][r], q2 = acc[f + 4][r];
                const size_t base = (((size_t)b * nHeads + h) * SEQ + pos) * DHEAD;
                outB[base + j]      = (__bf16)(q1 * cs - q2 * sn2);
                outB[base + j + 64] = (__bf16)(q2 * cs + q1 * sn2);
            }
        }
    } else if (MODE == 1) {     // transposed bf16 store: [b, h, dh, s]
        const int m_lo = m0 + hl * 8;
        const int b    = m_lo >> 11;
        const int pos0 = m_lo & (SEQ - 1);
#pragma unroll
        for (int f = 0; f < 8; ++f) {
            const int dh = f * 16 + l16;
            v8bf pk;
#pragma unroll
            for (int r = 0; r < 8; ++r) pk[r] = (__bf16)acc[f][r];
            *(v8bf*)&outB[(((size_t)b * nHeads + h) * DHEAD + dh) * SEQ + pos0] = pk;
        }
    } else {                    // fp32 row-major store
#pragma unroll
        for (int f = 0; f < 8; ++f) {
            const int n = n0 + f * 16 + l16;
#pragma unroll
            for (int r = 0; r < 8; ++r) {
                const int m = m0 + r + hl * 8;
                outF[(size_t)m * DMODEL + n] = acc[f][r];
            }
        }
    }
}

// ---------------------------------------------------------------------------
// Kernel 3: flash attention (causal, GQA).  grid = (S/128, B*HQ), block = 256.
//   wave tile: 16 queries x full head; K/V tiles of 64 keys double-buffered
//   in LDS via async global->LDS copies (8 x 16B per thread per tile).
// ---------------------------------------------------------------------------
__global__ __launch_bounds__(256, 1)
void flash_attn(const __bf16* __restrict__ Q, const __bf16* __restrict__ K,
                const __bf16* __restrict__ Vt, __bf16* __restrict__ Ctx) {
    __shared__ __bf16 Kl[2][64][128];       // keys  (32 KB)
    __shared__ __bf16 Vl[2][128][64];       // V^T   (32 KB)
    __shared__ __bf16 Pl[8][16][64];        // per-wave P staging (16 KB)

    const int lane = threadIdx.x & 31;
    const int wv   = threadIdx.x >> 5;
    const int l16  = lane & 15;
    const int hl   = lane >> 4;

    const int b  = blockIdx.y / HQ;
    const int h  = blockIdx.y % HQ;
    const int hk = h / GQA_G;
    const int q0 = blockIdx.x * 128;
    const int qw = q0 + wv * 16;            // wave's first query row

    const __bf16* Qb = Q  + ((size_t)b * HQ  + h)  * SEQ * DHEAD;
    const __bf16* Kb = K  + ((size_t)b * HKV + hk) * SEQ * DHEAD;
    const __bf16* Vb = Vt + ((size_t)b * HKV + hk) * DHEAD * SEQ;

    // staging roles
    const int kr = threadIdx.x >> 2, kseg = (threadIdx.x & 3) * 32;   // K tile
    const int vr = threadIdx.x >> 1, vseg = (threadIdx.x & 1) * 32;   // V tile

    // Q fragments for the whole head (4 chunks of K=32), loaded once
    v16bf aq[4];
    {
        const size_t qr = (size_t)(qw + l16) * DHEAD;
#pragma unroll
        for (int kd = 0; kd < 4; ++kd) {
            v8bf lo = *(const v8bf*)&Qb[qr + kd * 32 + hl * 8];
            v8bf hi = *(const v8bf*)&Qb[qr + kd * 32 + 16 + hl * 8];
            aq[kd]  = cat16(lo, hi);
        }
    }

    v8f o[8];
#pragma unroll
    for (int f = 0; f < 8; ++f) o[f] = (v8f){0,0,0,0,0,0,0,0};
    float mi[8], li[8];
#pragma unroll
    for (int r = 0; r < 8; ++r) { mi[r] = -3.0e38f; li[r] = 0.0f; }

    const float scale = 0.088388347648318447f;   // 1/sqrt(128)
    const int nT = q0 / 64 + 2;                  // cover keys 0..q0+127

    // prologue: stage KV tile 0 into buffer 0
    {
#pragma unroll
        for (int j = 0; j < 4; ++j)
            cp16_g2l(&Kl[0][kr][kseg + j * 8], &Kb[(size_t)kr * DHEAD + kseg + j * 8]);
#pragma unroll
        for (int j = 0; j < 4; ++j)
            cp16_g2l(&Vl[0][vr][vseg + j * 8], &Vb[(size_t)vr * SEQ + vseg + j * 8]);
    }

    for (int t = 0; t < nT; ++t) {
        const int cur = t & 1;
        const int k0  = t * 64;
        if (t + 1 < nT) {                   // issue next KV tile, drain current
            const int kn = k0 + 64;
#pragma unroll
            for (int j = 0; j < 4; ++j)
                cp16_g2l(&Kl[cur ^ 1][kr][kseg + j * 8],
                         &Kb[(size_t)(kn + kr) * DHEAD + kseg + j * 8]);
#pragma unroll
            for (int j = 0; j < 4; ++j)
                cp16_g2l(&Vl[cur ^ 1][vr][vseg + j * 8],
                         &Vb[(size_t)vr * SEQ + kn + vseg + j * 8]);
            ASYNC_WAIT(8);
        } else {
            ASYNC_WAIT(0);
        }
        __syncthreads();                    // tile t visible to all waves

        // ---- scores S = Q K^T  (16 queries x 64 keys) ----
        v8f sc[4];
#pragma unroll
        for (int nf = 0; nf < 4; ++nf) {
            sc[nf] = (v8f){0,0,0,0,0,0,0,0};
#pragma unroll
            for (int kd = 0; kd < 4; ++kd) {
                v8bf blo = *(const v8bf*)&Kl[cur][nf * 16 + l16][kd * 32 + hl * 8];
                v8bf bhi = *(const v8bf*)&Kl[cur][nf * 16 + l16][kd * 32 + 16 + hl * 8];
                sc[nf]   = wmma_bf16(aq[kd], cat16(blo, bhi), sc[nf]);
            }
        }

        // ---- causal mask + scale + online softmax ----
        float nm[8];
#pragma unroll
        for (int r = 0; r < 8; ++r) nm[r] = mi[r];
#pragma unroll
        for (int nf = 0; nf < 4; ++nf) {
            const int col = k0 + nf * 16 + l16;
#pragma unroll
            for (int r = 0; r < 8; ++r) {
                const int row = qw + r + hl * 8;
                float s = sc[nf][r] * scale;
                s = (col <= row) ? s : -3.0e38f;
                sc[nf][r] = s;
                nm[r] = fmaxf(nm[r], s);
            }
        }
#pragma unroll
        for (int r = 0; r < 8; ++r) {        // 16-lane row-max (stays in half)
            nm[r] = fmaxf(nm[r], __shfl_xor(nm[r], 1, 32));
            nm[r] = fmaxf(nm[r], __shfl_xor(nm[r], 2, 32));
            nm[r] = fmaxf(nm[r], __shfl_xor(nm[r], 4, 32));
            nm[r] = fmaxf(nm[r], __shfl_xor(nm[r], 8, 32));
        }
        float al[8], ps[8];
#pragma unroll
        for (int r = 0; r < 8; ++r) {
            al[r] = __expf(mi[r] - nm[r]);
            mi[r] = nm[r];
            ps[r] = 0.0f;
        }
#pragma unroll
        for (int nf = 0; nf < 4; ++nf)
#pragma unroll
            for (int r = 0; r < 8; ++r) {
                const float p = __expf(sc[nf][r] - nm[r]);
                sc[nf][r] = p;
                ps[r] += p;
            }
#pragma unroll
        for (int r = 0; r < 8; ++r) {        // 16-lane row-sum
            ps[r] += __shfl_xor(ps[r], 1, 32);
            ps[r] += __shfl_xor(ps[r], 2, 32);
            ps[r] += __shfl_xor(ps[r], 4, 32);
            ps[r] += __shfl_xor(ps[r], 8, 32);
            li[r] = li[r] * al[r] + ps[r];
        }
#pragma unroll
        for (int f = 0; f < 8; ++f)
#pragma unroll
            for (int r = 0; r < 8; ++r) o[f][r] *= al[r];

        // ---- P: C-layout -> LDS -> A-layout ----
#pragma unroll
        for (int nf = 0; nf < 4; ++nf)
#pragma unroll
            for (int r = 0; r < 8; ++r)
                Pl[wv][r + hl * 8][nf * 16 + l16] = (__bf16)sc[nf][r];
        __syncthreads();

        // ---- O += P V  (P: 16x64, V: 64x128) ----
#pragma unroll
        for (int kc = 0; kc < 2; ++kc) {
            v8bf plo = *(const v8bf*)&Pl[wv][l16][kc * 32 + hl * 8];
            v8bf phi = *(const v8bf*)&Pl[wv][l16][kc * 32 + 16 + hl * 8];
            v16bf ap = cat16(plo, phi);
#pragma unroll
            for (int f = 0; f < 8; ++f) {
                v8bf blo = *(const v8bf*)&Vl[cur][f * 16 + l16][kc * 32 + hl * 8];
                v8bf bhi = *(const v8bf*)&Vl[cur][f * 16 + l16][kc * 32 + 16 + hl * 8];
                o[f]     = wmma_bf16(ap, cat16(blo, bhi), o[f]);
            }
        }
        __syncthreads();                    // done reading buffers before overwrite
    }

    // ---- normalize + store context bf16 as [b, s, h*128+dh] ----
#pragma unroll
    for (int f = 0; f < 8; ++f) {
        const int dh = f * 16 + l16;
#pragma unroll
        for (int r = 0; r < 8; ++r) {
            const int pos = qw + r + hl * 8;
            const float v = o[f][r] / li[r];
            Ctx[((size_t)b * SEQ + pos) * DMODEL + h * DHEAD + dh] = (__bf16)v;
        }
    }
}

// ---------------------------------------------------------------------------
// Host launcher
// ---------------------------------------------------------------------------
extern "C" void kernel_launch(void* const* d_in, const int* in_sizes, int n_in,
                              void* d_out, int out_size, void* d_ws, size_t ws_size,
                              hipStream_t stream) {
    (void)n_in; (void)out_size; (void)ws_size;
    const float* x  = (const float*)d_in[0];
    const float* Wq = (const float*)d_in[1];
    const float* Wk = (const float*)d_in[2];
    const float* Wv = (const float*)d_in[3];
    const float* Wo = (const float*)d_in[4];

    char* ws = (char*)d_ws;
    // workspace layout (bytes), all 256-aligned; total ~76 MB
    __bf16* xb  = (__bf16*)(ws);                         // 4096*2048      = 16.8 MB
    __bf16* Wqb = (__bf16*)(ws + 16777216ull);           // 2048*2048      =  8.4 MB
    __bf16* Wkb = (__bf16*)(ws + 25165824ull);           //  512*2048      =  2.1 MB
    __bf16* Wvb = (__bf16*)(ws + 27262976ull);           //  512*2048      =  2.1 MB
    __bf16* Wob = (__bf16*)(ws + 29360128ull);           // 2048*2048      =  8.4 MB
    __bf16* Qr  = (__bf16*)(ws + 37748736ull);           // [2,16,2048,128]= 16.8 MB
    __bf16* Kr  = (__bf16*)(ws + 54525952ull);           // [2,4,2048,128] =  4.2 MB
    __bf16* Vt  = (__bf16*)(ws + 58720256ull);           // [2,4,128,2048] =  4.2 MB
    __bf16* Ctx = (__bf16*)(ws + 62914560ull);           // [4096,2048]    = 16.8 MB

    // 1) casts
    cast_to_bf16<<<1024, 256, 0, stream>>>(x,  xb,  in_sizes[0]);
    cast_to_bf16<<<1024, 256, 0, stream>>>(Wq, Wqb, in_sizes[1]);
    cast_to_bf16<<<1024, 256, 0, stream>>>(Wk, Wkb, in_sizes[2]);
    cast_to_bf16<<<1024, 256, 0, stream>>>(Wv, Wvb, in_sizes[3]);
    cast_to_bf16<<<1024, 256, 0, stream>>>(Wo, Wob, in_sizes[4]);

    // 2) projections (+ fused RoPE for Q/K, transposed store for V)
    gemm_wmma<0><<<dim3(MROWS / 128, (HQ  * DHEAD) / 128), 256, 0, stream>>>(xb, Wqb, Qr, nullptr, HQ);
    gemm_wmma<0><<<dim3(MROWS / 128, (HKV * DHEAD) / 128), 256, 0, stream>>>(xb, Wkb, Kr, nullptr, HKV);
    gemm_wmma<1><<<dim3(MROWS / 128, (HKV * DHEAD) / 128), 256, 0, stream>>>(xb, Wvb, Vt, nullptr, HKV);

    // 3) causal flash attention (GQA)
    flash_attn<<<dim3(SEQ / 128, BATCH * HQ), 256, 0, stream>>>(Qr, Kr, Vt, Ctx);

    // 4) output projection, fp32 result
    gemm_wmma<2><<<dim3(MROWS / 128, DMODEL / 128), 256, 0, stream>>>(Ctx, Wob, nullptr, (float*)d_out, 1);
}
